// ConditionalAutoregressive2D_37572373905628
// MI455X (gfx1250) — matrix operand order repaired and checked by reading, not compiled, not run on hardware
//
#include <hip/hip_runtime.h>
#include <hip/hip_bf16.h>
#include <math.h>
#include <stdint.h>

// ---- model dims (fixed by the reference) ----
#define NB     2
#define LSEQ   2048
#define WIDTH  1024
#define HEADS  4
#define DKH    64           // per-head dk == dv
#define HKD    256          // HEADS*DKH
#define FFD    1024
#define NTOK   (NB*LSEQ)    // 4096
#define BINS   2048
#define NMIX   10
#define DEPTH  4
#define CHUNK  128
#define NCHUNK (LSEQ/CHUNK) // 16

#define ACT_NONE 0
#define ACT_ELU1 1
#define ACT_GELU 2

// GEMM tiling: 64 rows x 128 cols per block, 8 waves, wave tile 16x64 (4 WMMA)
#define BM 64
#define BN 128

typedef __attribute__((ext_vector_type(16))) _Float16 v16h;
typedef __attribute__((ext_vector_type(8)))  _Float16 v8h;
typedef __attribute__((ext_vector_type(8)))  float    v8f;

__device__ __forceinline__ v16h cat8(v8h lo, v8h hi) {
  return __builtin_shufflevector(lo, hi, 0,1,2,3,4,5,6,7,8,9,10,11,12,13,14,15);
}

// CDNA5 async global->LDS copy (ASYNCcnt path). VDST = LDS byte address
// (low 32 bits of the flat LDS pointer per ISA 10.2 aperture rule).
__device__ __forceinline__ void async_copy_b128(void* ldsdst, const void* gsrc) {
  unsigned loff = (unsigned)(uintptr_t)ldsdst;
  asm volatile("global_load_async_to_lds_b128 %0, %1, off"
               :: "v"(loff), "v"((unsigned long long)(uintptr_t)gsrc)
               : "memory");
}
__device__ __forceinline__ void wait_async_zero() {
#if __has_builtin(__builtin_amdgcn_s_wait_asynccnt)
  __builtin_amdgcn_s_wait_asynccnt(0);
#else
  asm volatile("s_wait_asynccnt 0" ::: "memory");
#endif
}

// ---------------------------------------------------------------------------
// Weight prep: Wt[n][k] = (f16) W[k][n]   (32x32 LDS-tiled transpose+convert)
// ---------------------------------------------------------------------------
__global__ __launch_bounds__(256)
void wconvt_kernel(const float* __restrict__ W, _Float16* __restrict__ Wt,
                   int K, int Nc)
{
  __shared__ float tile[32][33];
  int kb = blockIdx.x * 32, nb = blockIdx.y * 32;
  int tx = threadIdx.x & 31, ty = threadIdx.x >> 5;  // 32 x 8
  #pragma unroll
  for (int j = 0; j < 32; j += 8) {
    int k = kb + ty + j, n = nb + tx;
    tile[ty + j][tx] = (n < Nc) ? W[(size_t)k * Nc + n] : 0.f;
  }
  __syncthreads();
  #pragma unroll
  for (int j = 0; j < 32; j += 8) {
    int n = nb + ty + j, k = kb + tx;
    if (n < Nc) Wt[(size_t)n * K + k] = (_Float16)tile[tx][ty + j];
  }
}

// ---------------------------------------------------------------------------
// Embedding: h[n,t,:] = (t==0 ? start : x_emb[x[n,t-1]]) + pos_emb[t]
// ---------------------------------------------------------------------------
__global__ __launch_bounds__(256)
void embed_kernel(const int* __restrict__ x, const float* __restrict__ x_emb,
                  const float* __restrict__ start_tok, const float* __restrict__ pos_emb,
                  float* __restrict__ h, _Float16* __restrict__ h16)
{
  int tok = blockIdx.x;
  int n = tok / LSEQ, t = tok % LSEQ;
  const float* src = (t == 0) ? start_tok
                              : (x_emb + (size_t)x[n * LSEQ + t - 1] * WIDTH);
  #pragma unroll
  for (int j = 0; j < WIDTH / 256; ++j) {
    int c = threadIdx.x + j * 256;
    float v = src[c] + pos_emb[(size_t)t * WIDTH + c];
    h[(size_t)tok * WIDTH + c]   = v;
    h16[(size_t)tok * WIDTH + c] = (_Float16)v;
  }
}

// ---------------------------------------------------------------------------
// Tiled WMMA GEMM: C = act(A16[MxK] @ Bt16[NcxK]^T + bias) (+ resid)
// f16 in / f32 accumulate; double-buffered async global->LDS pipeline.
// ---------------------------------------------------------------------------
__global__ __launch_bounds__(256)
void gemm_wmma(const _Float16* __restrict__ A, const _Float16* __restrict__ Bt,
               const float* __restrict__ bias, const float* __restrict__ resid,
               float* __restrict__ C, _Float16* __restrict__ C16,
               int M, int K, int Nc, int act)
{
  __shared__ __align__(16) _Float16 As[2][BM][40];   // double-buffered tiles
  __shared__ __align__(16) _Float16 Bs[2][BN][40];
  (void)M;
  int tid  = threadIdx.x;
  int m0   = blockIdx.y * BM;
  int n0   = blockIdx.x * BN;
  int wave = tid >> 5;
  int lane = tid & 31;
  int lrow = lane & 15;
  int lhalf= lane >> 4;
  int wm   = wave >> 1;   // 0..3  (16-row slab)
  int wn   = wave & 1;    // 0..1  (64-col slab)
  const bool fullN = (n0 + BN) <= Nc;   // block-uniform

  v8f acc0 = {}; v8f acc1 = {}; v8f acc2 = {}; v8f acc3 = {};

  // per-thread copy coordinates (16-byte chunks of f16)
  int arow = tid >> 2;                 // 0..63
  int akk  = (tid & 3) * 8;            // 0,8,16,24
  int bc0  = (tid * 2) >> 2;           // B chunk 0: col
  int bk0  = ((tid * 2) & 3) * 8;      //            kk
  int bc1  = (tid * 2 + 1) >> 2;       // B chunk 1: col
  int bk1  = ((tid * 2 + 1) & 3) * 8;  //            kk

  int r  = wm * 16 + lrow;             // A row for this lane
  int cb = wn * 64 + lrow;             // B column base for this lane

#define GEMM_COPY_TILE(buf, kk0)                                               \
  do {                                                                         \
    async_copy_b128(&As[buf][arow][akk],                                       \
                    A + (size_t)(m0 + arow) * K + ((kk0) + akk));              \
    async_copy_b128(&Bs[buf][bc0][bk0],                                        \
                    Bt + (size_t)(n0 + bc0) * K + ((kk0) + bk0));              \
    async_copy_b128(&Bs[buf][bc1][bk1],                                        \
                    Bt + (size_t)(n0 + bc1) * K + ((kk0) + bk1));              \
  } while (0)

#define GEMM_MMA_STEP(buf)                                                     \
  do {                                                                         \
    v8h alo = *(const v8h*)&As[buf][r][lhalf * 16];                            \
    v8h ahi = *(const v8h*)&As[buf][r][lhalf * 16 + 8];                        \
    v16h af = cat8(alo, ahi);                                                  \
    v8h b0lo = *(const v8h*)&Bs[buf][cb +  0][lhalf * 16];                     \
    v8h b0hi = *(const v8h*)&Bs[buf][cb +  0][lhalf * 16 + 8];                 \
    v8h b1lo = *(const v8h*)&Bs[buf][cb + 16][lhalf * 16];                     \
    v8h b1hi = *(const v8h*)&Bs[buf][cb + 16][lhalf * 16 + 8];                 \
    v8h b2lo = *(const v8h*)&Bs[buf][cb + 32][lhalf * 16];                     \
    v8h b2hi = *(const v8h*)&Bs[buf][cb + 32][lhalf * 16 + 8];                 \
    v8h b3lo = *(const v8h*)&Bs[buf][cb + 48][lhalf * 16];                     \
    v8h b3hi = *(const v8h*)&Bs[buf][cb + 48][lhalf * 16 + 8];                 \
    v16h bf0 = cat8(b0lo, b0hi);                                               \
    v16h bf1 = cat8(b1lo, b1hi);                                               \
    v16h bf2 = cat8(b2lo, b2hi);                                               \
    v16h bf3 = cat8(b3lo, b3hi);                                               \
    acc0 = __builtin_amdgcn_wmma_f32_16x16x32_f16(false, af, false, bf0,       \
                                                  (short)0, acc0, false, false);\
    acc1 = __builtin_amdgcn_wmma_f32_16x16x32_f16(false, af, false, bf1,       \
                                                  (short)0, acc1, false, false);\
    acc2 = __builtin_amdgcn_wmma_f32_16x16x32_f16(false, af, false, bf2,       \
                                                  (short)0, acc2, false, false);\
    acc3 = __builtin_amdgcn_wmma_f32_16x16x32_f16(false, af, false, bf3,       \
                                                  (short)0, acc3, false, false);\
  } while (0)

  if (fullN) {
    // ---- software-pipelined async path: prefetch k+1 while computing k ----
    GEMM_COPY_TILE(0, 0);
    wait_async_zero();
    __syncthreads();
    int cur = 0;
    for (int k0 = 0; k0 < K; k0 += 32) {
      if (k0 + 32 < K) GEMM_COPY_TILE(cur ^ 1, k0 + 32);
      GEMM_MMA_STEP(cur);
      wait_async_zero();
      __syncthreads();
      cur ^= 1;
    }
  } else {
    // ---- guarded synchronous path (only the Nc=30 logits GEMM) ----
    for (int k0 = 0; k0 < K; k0 += 32) {
      {
        v8h hv = *(const v8h*)(A + (size_t)(m0 + arow) * K + (k0 + akk));
        *(v8h*)&As[0][arow][akk] = hv;
      }
      #pragma unroll
      for (int cc = 0; cc < 2; ++cc) {
        int col = cc ? bc1 : bc0;
        int kk  = cc ? bk1 : bk0;
        int gcol = n0 + col;
        int ccol = (gcol < Nc) ? gcol : (Nc - 1);
        v8h hv = *(const v8h*)(Bt + (size_t)ccol * K + (k0 + kk));
        #pragma unroll
        for (int j = 0; j < 8; ++j) if (gcol >= Nc) hv[j] = (_Float16)0.f;
        *(v8h*)&Bs[0][col][kk] = hv;
      }
      __syncthreads();
      GEMM_MMA_STEP(0);
      __syncthreads();
    }
  }
#undef GEMM_COPY_TILE
#undef GEMM_MMA_STEP

  // store: VGPR v of C/D is row (lhalf*8 + v), col = lrow within 16-col tile
  #pragma unroll
  for (int g = 0; g < 4; ++g) {
    int col = n0 + wn * 64 + g * 16 + lrow;
    if (col >= Nc) continue;
    float bv = bias ? bias[col] : 0.f;
    #pragma unroll
    for (int v = 0; v < 8; ++v) {
      int row = m0 + wm * 16 + lhalf * 8 + v;
      float xv = (g == 0) ? acc0[v] : (g == 1) ? acc1[v] : (g == 2) ? acc2[v] : acc3[v];
      xv += bv;
      if (act == ACT_ELU1)      xv = (xv > 0.f) ? (xv + 1.f) : expf(xv);
      else if (act == ACT_GELU) xv = 0.5f * xv * (1.f + erff(xv * 0.70710678118654752f));
      if (resid) xv += resid[(size_t)row * Nc + col];
      if (C)   C  [(size_t)row * Nc + col] = xv;
      if (C16) C16[(size_t)row * Nc + col] = (_Float16)xv;
    }
  }
}

// ---------------------------------------------------------------------------
// LayerNorm over WIDTH, one block per token row; optional f32 and f16 outputs
// ---------------------------------------------------------------------------
__global__ __launch_bounds__(256)
void ln_kernel(const float* __restrict__ in, const float* __restrict__ g,
               const float* __restrict__ b, float* __restrict__ out,
               _Float16* __restrict__ out16)
{
  __shared__ float red[256];
  int row = blockIdx.x, tid = threadIdx.x;
  const float* xp = in + (size_t)row * WIDTH;
  float s = 0.f, s2 = 0.f;
  for (int c = tid; c < WIDTH; c += 256) { float v = xp[c]; s += v; s2 += v * v; }
  red[tid] = s; __syncthreads();
  for (int off = 128; off > 0; off >>= 1) { if (tid < off) red[tid] += red[tid + off]; __syncthreads(); }
  float mean = red[0] * (1.f / WIDTH);
  __syncthreads();
  red[tid] = s2; __syncthreads();
  for (int off = 128; off > 0; off >>= 1) { if (tid < off) red[tid] += red[tid + off]; __syncthreads(); }
  float var = red[0] * (1.f / WIDTH) - mean * mean;
  float inv = rsqrtf(var + 1e-5f);
  for (int c = tid; c < WIDTH; c += 256) {
    float v = (xp[c] - mean) * inv * g[c] + b[c];
    if (out)   out  [(size_t)row * WIDTH + c] = v;
    if (out16) out16[(size_t)row * WIDTH + c] = (_Float16)v;
  }
}

// ---------------------------------------------------------------------------
// Linear attention, chunk-parallel scan.
// Pass A: per-chunk sums of Kf^T V (64x64) and sum(Kf)
// ---------------------------------------------------------------------------
__global__ __launch_bounds__(256)
void attn_chunk_sum(const float* __restrict__ Kf, const float* __restrict__ Vv,
                    float* __restrict__ Ckv, float* __restrict__ Cks)
{
  __shared__ float kf_s[16][64];
  __shared__ float v_s[16][64];
  int b = blockIdx.x;
  int nh = b >> 4, c = b & (NCHUNK - 1);
  int n = nh >> 2, head = nh & 3;
  int tid = threadIdx.x;
  int dk = tid >> 2, dv0 = (tid & 3) * 16;   // thread owns kv[dk][dv0..dv0+15]
  float acc[16] = {0.f};
  float ks = 0.f;
  for (int t0 = 0; t0 < CHUNK; t0 += 16) {
    #pragma unroll
    for (int j = 0; j < 4; ++j) {
      int idx = tid + j * 256;
      int tt = idx >> 6, d = idx & 63;
      size_t base = ((size_t)(n * LSEQ + c * CHUNK + t0 + tt)) * HKD + head * DKH;
      kf_s[tt][d] = Kf[base + d];
      v_s[tt][d]  = Vv[base + d];
    }
    __syncthreads();
    for (int tt = 0; tt < 16; ++tt) {
      float kfv = kf_s[tt][dk];
      if ((tid & 3) == 0) ks += kfv;
      #pragma unroll
      for (int j = 0; j < 16; ++j) acc[j] += kfv * v_s[tt][dv0 + j];
    }
    __syncthreads();
  }
  float* outp = Ckv + (size_t)b * (DKH * DKH) + dk * DKH + dv0;
  #pragma unroll
  for (int j = 0; j < 16; ++j) outp[j] = acc[j];
  if ((tid & 3) == 0) Cks[(size_t)b * DKH + dk] = ks;
}

// Pass B: in-place exclusive prefix over chunks, one block per (n,head)
__global__ __launch_bounds__(256)
void attn_prefix(float* __restrict__ Ckv, float* __restrict__ Cks)
{
  int nh = blockIdx.x;
  int tid = threadIdx.x;
  int dk = tid >> 2, dv0 = (tid & 3) * 16;
  float run[16] = {0.f};
  float ksr = 0.f;
  for (int c = 0; c < NCHUNK; ++c) {
    float* p = Ckv + ((size_t)(nh * NCHUNK + c)) * (DKH * DKH) + dk * DKH + dv0;
    #pragma unroll
    for (int j = 0; j < 16; ++j) { float t = p[j]; p[j] = run[j]; run[j] += t; }
    if ((tid & 3) == 0) {
      float* pk = Cks + (size_t)(nh * NCHUNK + c) * DKH + dk;
      float t = *pk; *pk = ksr; ksr += t;
    }
  }
}

// Pass C: seed LDS state with the exclusive prefix, scan 128 tokens locally.
// Output written directly as f16 (feeds the Wo GEMM).
__global__ __launch_bounds__(256)
void attn_scan(const float* __restrict__ Qf, const float* __restrict__ Kf,
               const float* __restrict__ Vv, const float* __restrict__ Ckv,
               const float* __restrict__ Cks, _Float16* __restrict__ Att16)
{
  __shared__ float kv[DKH][DKH + 1];
  __shared__ float ksum[DKH];
  __shared__ float kf_s[DKH], vv_s[DKH], qf_s[DKH];
  __shared__ float red[4][DKH];
  __shared__ float dred[4];
  int b = blockIdx.x;
  int nh = b >> 4, c = b & (NCHUNK - 1);
  int n = nh >> 2, head = nh & 3;
  int tid = threadIdx.x;
  for (int idx = tid; idx < DKH * DKH; idx += 256)
    kv[idx >> 6][idx & 63] = Ckv[(size_t)b * (DKH * DKH) + idx];
  if (tid < DKH) ksum[tid] = Cks[(size_t)b * DKH + tid];
  __syncthreads();
  for (int t = 0; t < CHUNK; ++t) {
    size_t base = ((size_t)(n * LSEQ + c * CHUNK + t)) * HKD + head * DKH;
    if (tid < 64)       kf_s[tid]        = Kf[base + tid];
    else if (tid < 128) vv_s[tid - 64]   = Vv[base + tid - 64];
    else if (tid < 192) qf_s[tid - 128]  = Qf[base + tid - 128];
    __syncthreads();
    { // state update: kv += kf ⊗ v, ksum += kf
      int dk = tid >> 2, dv0 = (tid & 3) * 16;
      float kfv = kf_s[dk];
      #pragma unroll
      for (int j = 0; j < 16; ++j) kv[dk][dv0 + j] += kfv * vv_s[dv0 + j];
      if ((tid & 3) == 0) ksum[dk] += kfv;
    }
    __syncthreads();
    { // num[dv] = sum_dk qf[dk]*kv[dk][dv], split 4 ways over dk
      int dv = tid & 63, q = tid >> 6, dk0 = q * 16;
      float p = 0.f;
      #pragma unroll
      for (int j = 0; j < 16; ++j) p += qf_s[dk0 + j] * kv[dk0 + j][dv];
      red[q][dv] = p;
    }
    if (tid < 4) {
      float d = 0.f;
      #pragma unroll
      for (int j = 0; j < 16; ++j) d += qf_s[tid * 16 + j] * ksum[tid * 16 + j];
      dred[tid] = d;
    }
    __syncthreads();
    if (tid < 64) {
      float num = red[0][tid] + red[1][tid] + red[2][tid] + red[3][tid];
      float den = dred[0] + dred[1] + dred[2] + dred[3];
      Att16[base + tid] = (_Float16)(num / (den + 1e-6f));
    }
    __syncthreads();
  }
}

// ---------------------------------------------------------------------------
// DMOL loss per token, then tree-reduce to a scalar (mean / log 2)
// ---------------------------------------------------------------------------
__device__ __forceinline__ float softplusf_(float x) {
  if (x > 20.f)  return x;
  if (x < -20.f) return expf(x);
  return log1pf(expf(x));
}
__device__ __forceinline__ float sigmoidf_(float x) { return 1.f / (1.f + expf(-x)); }

__global__ __launch_bounds__(256)
void dmol_kernel(const float* __restrict__ logits, const int* __restrict__ x,
                 float* __restrict__ lossbuf)
{
  int tok = blockIdx.x * 256 + threadIdx.x;
  if (tok >= NTOK) return;
  const float* lp = logits + (size_t)tok * (3 * NMIX);
  float y  = 2.f * (float)x[tok] / (float)(BINS - 1) - 1.f;
  float hb = 1.f / (float)(BINS - 1);

  float mx = -1e30f;
  #pragma unroll
  for (int i = 0; i < NMIX; ++i) mx = fmaxf(mx, lp[i]);
  float se = 0.f;
  #pragma unroll
  for (int i = 0; i < NMIX; ++i) se += expf(lp[i] - mx);
  float lse = mx + logf(se);

  float terms[NMIX];
  float m2 = -1e30f;
  #pragma unroll
  for (int i = 0; i < NMIX; ++i) {
    float mean = lp[NMIX + i];
    float ls   = fmaxf(lp[2 * NMIX + i], -7.f);
    float cc   = y - mean;
    float inv  = expf(-ls);
    float plus_in = inv * (cc + hb);
    float min_in  = inv * (cc - hb);
    float cdf_delta = sigmoidf_(plus_in) - sigmoidf_(min_in);
    float lcp = plus_in - softplusf_(plus_in);
    float lom = -softplusf_(min_in);
    float mid = inv * cc;
    float lpm = mid - ls - 2.f * softplusf_(mid);
    float l;
    if (y < -0.999f)            l = lcp;
    else if (y > 0.999f)        l = lom;
    else if (cdf_delta > 1e-5f) l = logf(fmaxf(cdf_delta, 1e-12f));
    else                        l = lpm - logf((float)(BINS - 1) * 0.5f);
    l += lp[i] - lse;
    terms[i] = l;
    m2 = fmaxf(m2, l);
  }
  float s = 0.f;
  #pragma unroll
  for (int i = 0; i < NMIX; ++i) s += expf(terms[i] - m2);
  lossbuf[tok] = -(m2 + logf(s));
}

__global__ __launch_bounds__(256)
void loss_reduce(const float* __restrict__ lossbuf, float* __restrict__ out)
{
  __shared__ float red[256];
  float s = 0.f;
  for (int i = threadIdx.x; i < NTOK; i += 256) s += lossbuf[i];
  red[threadIdx.x] = s; __syncthreads();
  for (int off = 128; off > 0; off >>= 1) {
    if (threadIdx.x < off) red[threadIdx.x] += red[threadIdx.x + off];
    __syncthreads();
  }
  if (threadIdx.x == 0) out[0] = red[0] / ((float)NTOK * 0.6931471805599453f);
}

// ---------------------------------------------------------------------------
extern "C" void kernel_launch(void* const* d_in, const int* in_sizes, int n_in,
                              void* d_out, int out_size, void* d_ws, size_t ws_size,
                              hipStream_t stream)
{
  (void)in_sizes; (void)n_in; (void)out_size; (void)ws_size;
  const int*   x       = (const int*)d_in[0];
  const float* x_emb   = (const float*)d_in[1];
  const float* start_t = (const float*)d_in[2];
  const float* pos_emb = (const float*)d_in[3];
  const float* Wq  = (const float*)d_in[4];
  const float* bq  = (const float*)d_in[5];
  const float* Wk  = (const float*)d_in[6];
  const float* bk  = (const float*)d_in[7];
  const float* Wv  = (const float*)d_in[8];
  const float* bv  = (const float*)d_in[9];
  const float* Wo  = (const float*)d_in[10];
  const float* bo  = (const float*)d_in[11];
  const float* ln1g= (const float*)d_in[12];
  const float* ln1b= (const float*)d_in[13];
  const float* W1  = (const float*)d_in[14];
  const float* b1  = (const float*)d_in[15];
  const float* W2  = (const float*)d_in[16];
  const float* b2  = (const float*)d_in[17];
  const float* ln2g= (const float*)d_in[18];
  const float* ln2b= (const float*)d_in[19];
  const float* lnfg= (const float*)d_in[20];
  const float* lnfb= (const float*)d_in[21];
  const float* Wout= (const float*)d_in[22];

  // ---- workspace carve (byte based; all blocks 16B-aligned sizes) ----
  char* wsp = (char*)d_ws;
  auto allocF = [&](size_t n) { float*    p = (float*)wsp;    wsp += n * 4; return p; };
  auto allocH = [&](size_t n) { _Float16* p = (_Float16*)wsp; wsp += n * 2; return p; };

  float*    h     = allocF((size_t)NTOK * WIDTH);
  float*    Qf    = allocF((size_t)NTOK * HKD);
  float*    Kf    = allocF((size_t)NTOK * HKD);
  float*    Vv    = allocF((size_t)NTOK * HKD);
  float*    Tmp   = allocF((size_t)NTOK * WIDTH);
  float*    Ckv   = allocF((size_t)(NB * HEADS * NCHUNK) * DKH * DKH);
  float*    Cks   = allocF((size_t)(NB * HEADS * NCHUNK) * DKH);
  float*    Lg    = allocF((size_t)NTOK * (3 * NMIX));
  float*    Lb    = allocF((size_t)NTOK);
  _Float16* h16   = allocH((size_t)NTOK * WIDTH);
  _Float16* Att16 = allocH((size_t)NTOK * HKD);
  _Float16* T116  = allocH((size_t)NTOK * FFD);
  _Float16* Tmp16 = allocH((size_t)NTOK * WIDTH);
  _Float16* Wq_t  = allocH((size_t)DEPTH * HKD * WIDTH);
  _Float16* Wk_t  = allocH((size_t)DEPTH * HKD * WIDTH);
  _Float16* Wv_t  = allocH((size_t)DEPTH * HKD * WIDTH);
  _Float16* Wo_t  = allocH((size_t)DEPTH * WIDTH * HKD);
  _Float16* W1_t  = allocH((size_t)DEPTH * FFD * WIDTH);
  _Float16* W2_t  = allocH((size_t)DEPTH * WIDTH * FFD);
  _Float16* Wout_t= allocH((size_t)(3 * NMIX) * WIDTH);

  // ---- one-time (per launch) weight transpose + f16 convert ----
  for (int l = 0; l < DEPTH; ++l) {
    dim3 gqk(WIDTH / 32, HKD / 32);     // K=WIDTH, Nc=HKD
    wconvt_kernel<<<gqk, 256, 0, stream>>>(Wq + (size_t)l * WIDTH * HKD,
                                           Wq_t + (size_t)l * HKD * WIDTH, WIDTH, HKD);
    wconvt_kernel<<<gqk, 256, 0, stream>>>(Wk + (size_t)l * WIDTH * HKD,
                                           Wk_t + (size_t)l * HKD * WIDTH, WIDTH, HKD);
    wconvt_kernel<<<gqk, 256, 0, stream>>>(Wv + (size_t)l * WIDTH * HKD,
                                           Wv_t + (size_t)l * HKD * WIDTH, WIDTH, HKD);
    dim3 go(HKD / 32, WIDTH / 32);      // K=HKD, Nc=WIDTH
    wconvt_kernel<<<go, 256, 0, stream>>>(Wo + (size_t)l * HKD * WIDTH,
                                          Wo_t + (size_t)l * WIDTH * HKD, HKD, WIDTH);
    dim3 gff(WIDTH / 32, FFD / 32);     // square
    wconvt_kernel<<<gff, 256, 0, stream>>>(W1 + (size_t)l * WIDTH * FFD,
                                           W1_t + (size_t)l * FFD * WIDTH, WIDTH, FFD);
    wconvt_kernel<<<gff, 256, 0, stream>>>(W2 + (size_t)l * FFD * WIDTH,
                                           W2_t + (size_t)l * WIDTH * FFD, FFD, WIDTH);
  }
  {
    dim3 gwo(WIDTH / 32, (3 * NMIX + 31) / 32);   // K=WIDTH, Nc=30 (guarded)
    wconvt_kernel<<<gwo, 256, 0, stream>>>(Wout, Wout_t, WIDTH, 3 * NMIX);
  }

  embed_kernel<<<NTOK, 256, 0, stream>>>(x, x_emb, start_t, pos_emb, h, h16);

  dim3 gqkv(HKD / BN, NTOK / BM);     // (2, 64)
  dim3 gw(WIDTH / BN, NTOK / BM);     // (8, 64)
  dim3 gout(1, NTOK / BM);            // logits: Nc=30, guarded path

  for (int l = 0; l < DEPTH; ++l) {
    const _Float16* wq = Wq_t + (size_t)l * HKD * WIDTH;
    const _Float16* wk = Wk_t + (size_t)l * HKD * WIDTH;
    const _Float16* wv = Wv_t + (size_t)l * HKD * WIDTH;
    const _Float16* wo = Wo_t + (size_t)l * WIDTH * HKD;
    const _Float16* w1 = W1_t + (size_t)l * FFD * WIDTH;
    const _Float16* w2 = W2_t + (size_t)l * WIDTH * FFD;

    gemm_wmma<<<gqkv, 256, 0, stream>>>(h16, wq, bq + (size_t)l * HKD, nullptr,
                                        Qf, nullptr, NTOK, WIDTH, HKD, ACT_ELU1);
    gemm_wmma<<<gqkv, 256, 0, stream>>>(h16, wk, bk + (size_t)l * HKD, nullptr,
                                        Kf, nullptr, NTOK, WIDTH, HKD, ACT_ELU1);
    gemm_wmma<<<gqkv, 256, 0, stream>>>(h16, wv, bv + (size_t)l * HKD, nullptr,
                                        Vv, nullptr, NTOK, WIDTH, HKD, ACT_NONE);

    attn_chunk_sum<<<NB * HEADS * NCHUNK, 256, 0, stream>>>(Kf, Vv, Ckv, Cks);
    attn_prefix  <<<NB * HEADS,          256, 0, stream>>>(Ckv, Cks);
    attn_scan    <<<NB * HEADS * NCHUNK, 256, 0, stream>>>(Qf, Kf, Vv, Ckv, Cks, Att16);

    gemm_wmma<<<gw, 256, 0, stream>>>(Att16, wo, bo + (size_t)l * WIDTH, h,
                                      Tmp, nullptr, NTOK, HKD, WIDTH, ACT_NONE);
    ln_kernel<<<NTOK, 256, 0, stream>>>(Tmp, ln1g + (size_t)l * WIDTH,
                                        ln1b + (size_t)l * WIDTH, h, h16);
    gemm_wmma<<<gw, 256, 0, stream>>>(h16, w1, b1 + (size_t)l * FFD, nullptr,
                                      nullptr, T116, NTOK, WIDTH, FFD, ACT_GELU);
    gemm_wmma<<<gw, 256, 0, stream>>>(T116, w2, b2 + (size_t)l * WIDTH, h,
                                      Tmp, nullptr, NTOK, FFD, WIDTH, ACT_NONE);
    ln_kernel<<<NTOK, 256, 0, stream>>>(Tmp, ln2g + (size_t)l * WIDTH,
                                        ln2b + (size_t)l * WIDTH, h, h16);
  }

  ln_kernel<<<NTOK, 256, 0, stream>>>(h, lnfg, lnfb, nullptr, Tmp16);
  gemm_wmma<<<gout, 256, 0, stream>>>(Tmp16, Wout_t, nullptr, nullptr,
                                      Lg, nullptr, NTOK, WIDTH, 3 * NMIX, ACT_NONE);
  dmol_kernel<<<NTOK / 256, 256, 0, stream>>>(Lg, x, Lb);
  loss_reduce<<<1, 256, 0, stream>>>(Lb, (float*)d_out);
}
